// TopoGradLoss_88459146428758
// MI455X (gfx1250) — matrix operand order repaired and checked by reading, not compiled, not run on hardware
//
#include <hip/hip_runtime.h>
#include <hip/hip_bf16.h>
#include <math.h>

#define NPTS 8192
#define DIMS 64
#define KKDE 50
#define KRIPS 16
#define SCALE_F 30.0f
#define DESTNUM 10
#define THRESHOLD_F 1.0f
#define NBINS 128
#define DMAX 512.0f

// B-panel staging geometry: 8 column tiles (8*16 rows * 128B/row = 16KB) per panel
#define PTILES 8
#define PROWS (PTILES * 16)                 // 128 columns per panel
#define PANEL_H (PROWS * DIMS)              // halves per panel = 8192 (16KB)
#define PANEL_BYTES (PANEL_H * 2)           // 16384
#define NPANELS (NPTS / PROWS)              // 64

typedef __attribute__((ext_vector_type(16))) _Float16 v16h;
typedef __attribute__((ext_vector_type(8)))  _Float16 v8h;
typedef __attribute__((ext_vector_type(8)))  float    v8f;
typedef __attribute__((ext_vector_type(4)))  unsigned u32x4;
typedef __attribute__((ext_vector_type(8)))  unsigned u32x8;

// ---- WMMA fragment loaders (contiguous-K A/B order, same in A and B, CDNA5 ISA §7.12.2) ----
// A (16x32 f16): lane<16 -> row=lane, K 0..15; lane>=16 -> row=lane-16, K 16..31
__device__ __forceinline__ v16h load_a_frag(const _Float16* xh, int row, int lane, int chunk) {
  const int kb = (lane < 16) ? 0 : 16;
  const _Float16* p = xh + (size_t)row * DIMS + chunk * 32 + kb;
  v8h lo = *(const v8h*)p;
  v8h hi = *(const v8h*)(p + 8);
  v16h r;
#pragma unroll
  for (int t = 0; t < 8; ++t) { r[t] = lo[t]; r[t + 8] = hi[t]; }
  return r;
}
// B (32x16 f16) fragment out of an LDS-resident panel: lane%16 = column n,
// lane<16 holds K 0..15, lane>=16 holds K 16..31 (contiguous halves of row j0+n)
__device__ __forceinline__ v16h lds_b_frag(const _Float16* panel_buf, int tt, int lane, int chunk) {
  const int n  = lane & 15;
  const int ko = (lane < 16) ? 0 : 16;
  return *(const v16h*)(panel_buf + (size_t)(tt * 16 + n) * DIMS + chunk * 32 + ko);
}

// ---- CDNA5 async global->LDS panel stage (ASYNCcnt path, CDNA5 ISA op 98) ----
// 128 threads x 8 instructions x 16B = 16KB panel; per-wave in-order completion.
__device__ __forceinline__ void async_stage_panel(const _Float16* xh, int p, _Float16* panel_buf, int tid) {
  unsigned lbase = (unsigned)(uintptr_t)panel_buf;                  // flat addr low 32b = LDS offset
  unsigned long long gbase =
      (unsigned long long)(uintptr_t)xh + (unsigned long long)p * PANEL_BYTES;
#pragma unroll
  for (int t = 0; t < 8; ++t) {
    unsigned off = (unsigned)(tid * 16 + t * 2048);
    asm volatile("global_load_async_to_lds_b128 %0, %1, off"
                 :: "v"(lbase + off), "v"(gbase + (unsigned long long)off)
                 : "memory");
  }
}
// Column-norm slice for the panel: 128 floats = 512B, staged by wave 0 only.
__device__ __forceinline__ void async_stage_sq(const float* sqn, int p, float* sq_buf, int tid) {
  if (tid < 32) {
    unsigned l = (unsigned)(uintptr_t)sq_buf + (unsigned)(tid * 16);
    unsigned long long g =
        (unsigned long long)(uintptr_t)(sqn + (size_t)p * PROWS) + (unsigned long long)(tid * 16);
    asm volatile("global_load_async_to_lds_b128 %0, %1, off" :: "v"(l), "v"(g) : "memory");
  }
}
__device__ __forceinline__ void wait_async0() {
  asm volatile("s_wait_asynccnt 0x0" ::: "memory");
}

// ---- CDNA5 Tensor Data Mover 1-D tile load (TENSORcnt path, CDNA5 ISA-8) ----
// D# group1: data_size=4B, tile = til elems x 1 row, tensor length tot elems.
__device__ __forceinline__ u32x8 tdm_group1(unsigned tot, unsigned til) {
  u32x8 g1;
  g1[0] = (2u << 16);                  // workgroup_mask=0 | data_size=2 (4B)
  g1[1] = (tot & 0xffffu) << 16;       // tensor_dim0[15:0]
  g1[2] = (tot >> 16) | (1u << 16);    // tensor_dim0[31:16] | tensor_dim1[15:0]=1
  g1[3] = (til << 16);                 // tensor_dim1[31:16]=0 | tile_dim0
  g1[4] = 1u;                          // tile_dim1=1 | tile_dim2=0
  g1[5] = tot;                         // tensor_dim0_stride[31:0]
  g1[6] = 0u;                          // stride0[47:32] | stride1[15:0]
  g1[7] = 0u;                          // stride1[47:16]
  return g1;
}
__device__ __forceinline__ void tdm_stage(const void* gsrc, void* lds_buf, u32x8 g1) {
  unsigned long long gaddr = (unsigned long long)(uintptr_t)gsrc;
  u32x4 g0;
  g0[0] = 1u;                                            // count=1, user-mode D#
  g0[1] = (unsigned)(uintptr_t)lds_buf;                  // lds_addr
  g0[2] = (unsigned)(gaddr & 0xffffffffu);               // global_addr[31:0]
  g0[3] = (unsigned)((gaddr >> 32) & 0x01ffffffu) | 0x80000000u;  // global_addr[56:32] | type=2
  asm volatile("tensor_load_to_lds %0, %1" :: "s"(g0), "s"(g1) : "memory");
}

// ---- 1) f32 -> f16 copy + squared norms ----
__global__ __launch_bounds__(256) void prep_kernel(const float* __restrict__ x,
                                                   _Float16* __restrict__ xh,
                                                   float* __restrict__ sqn) {
  int i = blockIdx.x * blockDim.x + threadIdx.x;
  if (i >= NPTS) return;
  const float* r = x + (size_t)i * DIMS;
  _Float16* o = xh + (size_t)i * DIMS;
  float s = 0.f;
#pragma unroll 8
  for (int d = 0; d < DIMS; ++d) { float v = r[d]; s += v * v; o[d] = (_Float16)v; }
  sqn[i] = s;
}

// ---- 2) KDE density: async-staged WMMA Gram tiles + histogram select + exp-sum ----
__global__ __launch_bounds__(128) void kde_kernel(const _Float16* __restrict__ xh,
                                                  const float* __restrict__ sqn,
                                                  float* __restrict__ res) {
  __shared__ __align__(128) _Float16 panel[2][PANEL_H];   // 2 x 16KB double buffer
  __shared__ __align__(64) float sq_panel[2][PROWS];      // column norms per panel
  __shared__ int   hist[16][NBINS];
  __shared__ float thresh[16];
  __shared__ float rowsum[16];
  __shared__ float sqrow[16];
  const int i0   = blockIdx.x * 16;
  const int wave = threadIdx.x >> 5;
  const int lane = threadIdx.x & 31;

  for (int t = threadIdx.x; t < 16 * NBINS; t += blockDim.x) (&hist[0][0])[t] = 0;
  if (threadIdx.x < 16) { rowsum[threadIdx.x] = 0.f; sqrow[threadIdx.x] = sqn[i0 + threadIdx.x]; }
  __syncthreads();

  const int row   = lane & 15;
  const int mbase = (lane < 16) ? 0 : 8;
  v16h a0 = load_a_frag(xh, i0 + row, lane, 0);
  v16h a1 = load_a_frag(xh, i0 + row, lane, 1);

  // ---- pass 1: histogram of squared distances ----
  async_stage_panel(xh, 0, &panel[0][0], threadIdx.x);
  async_stage_sq(sqn, 0, &sq_panel[0][0], threadIdx.x);
  wait_async0();
  __syncthreads();
  for (int p = 0; p < NPANELS; ++p) {
    const int b = p & 1;
    if (p + 1 < NPANELS) {
      async_stage_panel(xh, p + 1, &panel[b ^ 1][0], threadIdx.x);
      async_stage_sq(sqn, p + 1, &sq_panel[b ^ 1][0], threadIdx.x);
    }
    for (int tt = wave; tt < PTILES; tt += 4) {
      v16h b0 = lds_b_frag(&panel[b][0], tt, lane, 0);
      v16h b1 = lds_b_frag(&panel[b][0], tt, lane, 1);
      v8f c = {};
      c = __builtin_amdgcn_wmma_f32_16x16x32_f16(false, a0, false, b0, (short)0, c, false, false);
      c = __builtin_amdgcn_wmma_f32_16x16x32_f16(false, a1, false, b1, (short)0, c, false, false);
      const float sb = sq_panel[b][tt * 16 + (lane & 15)];
#pragma unroll
      for (int v = 0; v < 8; ++v) {
        const int m = v + mbase;
        float d = sqrow[m] + sb - 2.0f * c[v];
        int bi = (int)(d * ((float)NBINS / DMAX));
        bi = bi < 0 ? 0 : (bi > NBINS - 1 ? NBINS - 1 : bi);
        atomicAdd(&hist[m][bi], 1);
      }
    }
    wait_async0();
    __syncthreads();
  }
  if (threadIdx.x < 16) {
    int cnt = 0, bb = 0;
    while (bb < NBINS) { cnt += hist[threadIdx.x][bb]; if (cnt >= KKDE) break; ++bb; }
    thresh[threadIdx.x] = (float)(bb + 1) * (DMAX / (float)NBINS);
  }
  __syncthreads();

  // ---- pass 2: exp-sum below threshold ----
  async_stage_panel(xh, 0, &panel[0][0], threadIdx.x);
  async_stage_sq(sqn, 0, &sq_panel[0][0], threadIdx.x);
  wait_async0();
  __syncthreads();
  for (int p = 0; p < NPANELS; ++p) {
    const int b = p & 1;
    if (p + 1 < NPANELS) {
      async_stage_panel(xh, p + 1, &panel[b ^ 1][0], threadIdx.x);
      async_stage_sq(sqn, p + 1, &sq_panel[b ^ 1][0], threadIdx.x);
    }
    for (int tt = wave; tt < PTILES; tt += 4) {
      v16h b0 = lds_b_frag(&panel[b][0], tt, lane, 0);
      v16h b1 = lds_b_frag(&panel[b][0], tt, lane, 1);
      v8f c = {};
      c = __builtin_amdgcn_wmma_f32_16x16x32_f16(false, a0, false, b0, (short)0, c, false, false);
      c = __builtin_amdgcn_wmma_f32_16x16x32_f16(false, a1, false, b1, (short)0, c, false, false);
      const float sb = sq_panel[b][tt * 16 + (lane & 15)];
#pragma unroll
      for (int v = 0; v < 8; ++v) {
        const int m = v + mbase;
        float d = sqrow[m] + sb - 2.0f * c[v];
        if (d <= thresh[m]) atomicAdd(&rowsum[m], __expf(-d * (1.0f / SCALE_F)));
      }
    }
    wait_async0();
    __syncthreads();
  }
  if (threadIdx.x < 16)
    res[i0 + threadIdx.x] = rowsum[threadIdx.x] * (1.0f / ((float)KKDE * SCALE_F));
}

// ---- 3) max reduction ----
__global__ __launch_bounds__(1024) void maxred_kernel(const float* __restrict__ res, float* __restrict__ maxv) {
  __shared__ float sm[1024];
  float m = 0.f;
  for (int i = threadIdx.x; i < NPTS; i += 1024) m = fmaxf(m, res[i]);
  sm[threadIdx.x] = m;
  __syncthreads();
  for (int s = 512; s > 0; s >>= 1) {
    if (threadIdx.x < s) sm[threadIdx.x] = fmaxf(sm[threadIdx.x], sm[threadIdx.x + s]);
    __syncthreads();
  }
  if (threadIdx.x == 0) *maxv = sm[0];
}

// ---- 4) normalize + init sort payload ----
__global__ __launch_bounds__(256) void norm_init_kernel(const float* __restrict__ res, const float* __restrict__ maxv,
                                                        float* __restrict__ key, int* __restrict__ idx) {
  int i = blockIdx.x * blockDim.x + threadIdx.x;
  if (i >= NPTS) return;
  key[i] = res[i] / maxv[0];
  idx[i] = i;
}

// ---- 5) bitonic sort stage (ascending density) ----
__global__ __launch_bounds__(256) void bitonic_kernel(float* __restrict__ key, int* __restrict__ val, int k, int j) {
  int i = blockIdx.x * blockDim.x + threadIdx.x;
  int ixj = i ^ j;
  if (ixj > i) {
    float ki = key[i], kx = key[ixj];
    bool up = ((i & k) == 0);
    if ((up && ki > kx) || (!up && ki < kx)) {
      key[i] = kx; key[ixj] = ki;
      int vi = val[i]; val[i] = val[ixj]; val[ixj] = vi;
    }
  }
}

// ---- 6) gather sorted points ----
__global__ __launch_bounds__(256) void gather_kernel(const _Float16* __restrict__ xh, const float* __restrict__ sqn,
                                                     const int* __restrict__ idx,
                                                     _Float16* __restrict__ xsh, float* __restrict__ sqns) {
  int i = blockIdx.x * blockDim.x + threadIdx.x;
  if (i >= NPTS) return;
  int s = idx[i];
#pragma unroll 8
  for (int d = 0; d < DIMS; ++d) xsh[(size_t)i * DIMS + d] = xh[(size_t)s * DIMS + d];
  sqns[i] = sqn[s];
}

// ---- 7) rips 16-NN: TDM-staged WMMA tiles + per-row top-16 lists in LDS ----
__global__ __launch_bounds__(32) void rips_kernel(const _Float16* __restrict__ xh, const float* __restrict__ sqn,
                                                  int* __restrict__ knn) {
  __shared__ __align__(128) _Float16 panel[2][PANEL_H];
  __shared__ __align__(64) float sq_panel[2][PROWS];
  __shared__ float dt[16][16];
  __shared__ float bestd[16][KRIPS];
  __shared__ int   besti[16][KRIPS];
  __shared__ float sqrow[16];
  const int lane = threadIdx.x;
  const int i0 = blockIdx.x * 16;
  if (lane < 16) {
    sqrow[lane] = sqn[i0 + lane];
    for (int t = 0; t < KRIPS; ++t) { bestd[lane][t] = 3.4e38f; besti[lane][t] = 0; }
  }
  __syncthreads();
  const int row   = lane & 15;
  const int mbase = (lane < 16) ? 0 : 8;
  v16h a0 = load_a_frag(xh, i0 + row, lane, 0);
  v16h a1 = load_a_frag(xh, i0 + row, lane, 1);

  const u32x8 g1p = tdm_group1((unsigned)NPTS * DIMS * 2 / 4, PANEL_BYTES / 4);  // x panel
  const u32x8 g1s = tdm_group1((unsigned)NPTS, PROWS);                           // sqn slice
  tdm_stage((const char*)xh, &panel[0][0], g1p);
  tdm_stage(sqn, &sq_panel[0][0], g1s);
  __builtin_amdgcn_s_wait_tensorcnt(0);
  __syncthreads();
  for (int p = 0; p < NPANELS; ++p) {
    const int b = p & 1;
    if (p + 1 < NPANELS) {
      tdm_stage((const char*)xh + (size_t)(p + 1) * PANEL_BYTES, &panel[b ^ 1][0], g1p);
      tdm_stage(sqn + (size_t)(p + 1) * PROWS, &sq_panel[b ^ 1][0], g1s);
    }
    for (int tt = 0; tt < PTILES; ++tt) {
      const int j0 = p * PROWS + tt * 16;
      v16h b0 = lds_b_frag(&panel[b][0], tt, lane, 0);
      v16h b1 = lds_b_frag(&panel[b][0], tt, lane, 1);
      v8f c = {};
      c = __builtin_amdgcn_wmma_f32_16x16x32_f16(false, a0, false, b0, (short)0, c, false, false);
      c = __builtin_amdgcn_wmma_f32_16x16x32_f16(false, a1, false, b1, (short)0, c, false, false);
      const float sb = sq_panel[b][tt * 16 + (lane & 15)];
#pragma unroll
      for (int v = 0; v < 8; ++v) {
        const int m = v + mbase;
        dt[m][lane & 15] = sqrow[m] + sb - 2.0f * c[v];
      }
      __syncthreads();
      if (lane < 16) {
        for (int n = 0; n < 16; ++n) {
          float d = dt[lane][n];
          if (d < bestd[lane][KRIPS - 1]) {
            int q = KRIPS - 1;
            while (q > 0 && bestd[lane][q - 1] > d) {
              bestd[lane][q] = bestd[lane][q - 1];
              besti[lane][q] = besti[lane][q - 1];
              --q;
            }
            bestd[lane][q] = d; besti[lane][q] = j0 + n;
          }
        }
      }
      __syncthreads();
    }
    if (p + 1 < NPANELS) __builtin_amdgcn_s_wait_tensorcnt(0);
    __syncthreads();
  }
  if (lane < 16)
    for (int t = 0; t < KRIPS; ++t) knn[(size_t)(i0 + lane) * KRIPS + t] = besti[lane][t];
}

// ---- 8) sequential union-find persistence clustering (inherently serial) ----
__device__ int uf_find(int* parent, int i) {
  int root = i;
  while (parent[root] != root) root = parent[root];
  while (parent[i] != root) { int nxt = parent[i]; parent[i] = root; i = nxt; }
  return root;
}

__global__ void cluster_kernel(const float* __restrict__ dens_s, const int* __restrict__ rips,
                               int* __restrict__ parent, int* __restrict__ pairs, int* __restrict__ npairs) {
  for (int t = 0; t < NPTS; ++t) parent[t] = t;
  int np = 0;
  for (int i = NPTS - 2; i >= 0; --i) {
    int mx = -1;
    for (int t = 0; t < KRIPS; ++t) {
      int j = rips[(size_t)i * KRIPS + t];
      if (j > i && j > mx) mx = j;
    }
    if (mx < 0) continue;
    int eup = uf_find(parent, mx);                 // name[] is identity in the reference
    parent[uf_find(parent, i)] = eup;
    for (int t = 0; t < KRIPS; ++t) {
      int j = rips[(size_t)i * KRIPS + t];
      if (j <= i) continue;
      int ei = uf_find(parent, j);
      if (ei != eup) {
        pairs[2 * np] = ei; pairs[2 * np + 1] = i; ++np;
        if (dens_s[ei] - dens_s[i] < THRESHOLD_F)
          parent[uf_find(parent, ei)] = uf_find(parent, eup);
      }
    }
  }
  *npairs = np;
}

// ---- 9) PD pairs + top-10 persistence + final scalar loss ----
__global__ void loss_kernel(const float* __restrict__ dens_s, const int* __restrict__ pairs,
                            const int* __restrict__ npairs, int* __restrict__ maxref,
                            float* __restrict__ pers, float* __restrict__ out) {
  const int P = *npairs;
  for (int i = 0; i < NPTS; ++i) { maxref[i] = -1; pers[i] = -3.4e38f; }
  for (int p = 0; p < P; ++p) {
    int e = pairs[2 * p], r = pairs[2 * p + 1];
    if (r > maxref[e]) maxref[e] = r;
  }
  float sumall = 0.f;
  for (int e = 0; e < NPTS; ++e)
    if (maxref[e] >= 0) { float pv = dens_s[e] - dens_s[maxref[e]]; pers[e] = pv; sumall += pv; }
  int top_e[DESTNUM]; float top_p[DESTNUM];
  for (int t = 0; t < DESTNUM; ++t) {
    int be = -1; float bp = -3.4e38f;
    for (int e = 0; e < NPTS; ++e) if (pers[e] > bp) { bp = pers[e]; be = e; }
    top_e[t] = be; top_p[t] = bp;
    if (be >= 0) pers[be] = -3.4e38f;
  }
  float weak = sumall;
  for (int t = 0; t < DESTNUM; ++t) if (top_e[t] >= 0) weak -= top_p[t];
  weak *= 0.70710678118654752f;                    // /sqrt(2)
  float dest0 = 0.f, dest1 = 0.f;
  if (top_e[0] >= 0) { dest0 = dens_s[top_e[0]]; dest1 = dens_s[maxref[top_e[0]]]; }
  float strong = 0.f;
  for (int t = 1; t < DESTNUM; ++t)
    if (top_e[t] >= 0) {
      float d0 = dens_s[top_e[t]] - dest0;
      float d1 = dens_s[maxref[top_e[t]]] - dest1;
      strong += sqrtf(d0 * d0 + d1 * d1);
    }
  out[0] = weak + strong;
}

extern "C" void kernel_launch(void* const* d_in, const int* in_sizes, int n_in,
                              void* d_out, int out_size, void* d_ws, size_t ws_size,
                              hipStream_t stream) {
  (void)in_sizes; (void)n_in; (void)out_size; (void)ws_size;
  const float* x = (const float*)d_in[0];
  float* out = (float*)d_out;

  char* w = (char*)d_ws;
  size_t off = 0;
  auto take = [&](size_t bytes) -> void* {
    void* p = w + off;
    off = (off + bytes + 255) & ~(size_t)255;
    return p;
  };
  _Float16* xh   = (_Float16*)take((size_t)NPTS * DIMS * sizeof(_Float16));
  float*    sqn  = (float*)take((size_t)NPTS * sizeof(float));
  float*    res  = (float*)take((size_t)NPTS * sizeof(float));
  float*    maxv = (float*)take(256);
  float*    key  = (float*)take((size_t)NPTS * sizeof(float));   // becomes dens_s after sort
  int*      idx  = (int*)take((size_t)NPTS * sizeof(int));
  _Float16* xsh  = (_Float16*)take((size_t)NPTS * DIMS * sizeof(_Float16));
  float*    sqns = (float*)take((size_t)NPTS * sizeof(float));
  int*      ripsi = (int*)take((size_t)NPTS * KRIPS * sizeof(int));
  int*      parent = (int*)take((size_t)NPTS * sizeof(int));
  int*      pairs  = (int*)take((size_t)NPTS * KRIPS * 2 * sizeof(int));
  int*      npairs = (int*)take(256);
  int*      maxref = (int*)take((size_t)NPTS * sizeof(int));
  float*    pers   = (float*)take((size_t)NPTS * sizeof(float));

  prep_kernel<<<NPTS / 256, 256, 0, stream>>>(x, xh, sqn);
  kde_kernel<<<NPTS / 16, 128, 0, stream>>>(xh, sqn, res);
  maxred_kernel<<<1, 1024, 0, stream>>>(res, maxv);
  norm_init_kernel<<<NPTS / 256, 256, 0, stream>>>(res, maxv, key, idx);
  for (int k = 2; k <= NPTS; k <<= 1)
    for (int j = k >> 1; j > 0; j >>= 1)
      bitonic_kernel<<<NPTS / 256, 256, 0, stream>>>(key, idx, k, j);
  gather_kernel<<<NPTS / 256, 256, 0, stream>>>(xh, sqn, idx, xsh, sqns);
  rips_kernel<<<NPTS / 16, 32, 0, stream>>>(xsh, sqns, ripsi);
  cluster_kernel<<<1, 1, 0, stream>>>(key, ripsi, parent, pairs, npairs);
  loss_kernel<<<1, 1, 0, stream>>>(key, pairs, npairs, maxref, pers, out);
}
